// AttentionGraphSAGE_82454782148684
// MI455X (gfx1250) — compile-verified, matrix operand order checked by reading, not run on hardware
//
#include <hip/hip_runtime.h>
#include <hip/hip_bf16.h>

// ---------------------------------------------------------------------------
// AttentionGraphSAGE on MI455X (gfx1250), wave32 + v_wmma_f32_16x16x32_f16.
//
// Pipeline (all on `stream`):
//   0) memset agg/sums, convert x -> f16, repack W1/conv_w0/conv_w1 into
//      WMMA B-fragment order (per-lane 16 contiguous halves).
//   1) attn_kernel: per wave = 16 edges. Gather ef=[x[row]||x[col]] (f16) to
//      LDS, 4 heads x 8 N-tiles x 4 K-steps of WMMA, branch-free tanh
//      epilogue (v_exp_f32 + v_rcp_f32), dot with w2 via cross-lane shfl
//      reduction, store scores + hierarchical exp-sum.
//   2) wnorm_kernel: scores <- exp(scores)/sum[h]  (global softmax, in place)
//   3) scatter_kernel: agg[row] += w*x[col] via f32 global atomics (L2-resident)
//   4) node_kernel: per wave = 16 nodes, fused agg@W0 -> BN -> relu ->
//      @W1 -> BN -> relu -> @cls_w, all GEMMs via WMMA.
// ---------------------------------------------------------------------------

typedef __attribute__((ext_vector_type(16))) _Float16 v16h;
typedef __attribute__((ext_vector_type(8)))  _Float16 v8h;
typedef __attribute__((ext_vector_type(8)))  float    v8f;

union V16 { v16h v; struct { v8h lo, hi; } p; };

#define HEADS 4
#define IN    64
#define HID   128
#define EPS   1e-5f

// Branch-free tanh: 1 - 2/(e^{2x}+1). Saturates to +/-1 for large |x|
// (e^{2x} -> inf gives 1-0, -> 0 gives -1). Lowers to v_exp_f32 + v_rcp_f32.
__device__ __forceinline__ float fast_tanh(float x) {
  float t = __expf(2.f * x);
  return 1.f - 2.f * __builtin_amdgcn_rcpf(t + 1.f);
}

// WMMA f16 A-fragment (16x32): lane L holds M=L%16;
//   a[i], i<8 : K = 32*t + (L>=16?8:0) + i
//   a[i], i>=8: K = 32*t + 16 + (L>=16?8:0) + (i-8)
// B-fragment mirrors this with N=L%16 along K.
// C/D (16x16 f32): c[r] = (M = r + (L>=16?8:0), N = L%16).

// ---------------- prep kernels ----------------
__global__ void cvt_f16_kernel(const float* __restrict__ src,
                               _Float16* __restrict__ dst, int n) {
  int gid = blockIdx.x * blockDim.x + threadIdx.x;
  if (gid < n) dst[gid] = (_Float16)src[gid];
}

// Repack a row-major f32 W[K][Ncols] into B-fragment order:
// out[((t*NT + j)*32 + lane)*16 + i]  (KT = K/32, NT = Ncols/16)
__global__ void pack_b_kernel(const float* __restrict__ W,
                              _Float16* __restrict__ out,
                              int NT, int Ncols, int total) {
  int gid = blockIdx.x * blockDim.x + threadIdx.x;
  if (gid >= total) return;
  int i    = gid & 15;
  int lane = (gid >> 4) & 31;
  int j    = (gid >> 9) % NT;
  int t    = gid / (512 * NT);
  int n    = j * 16 + (lane & 15);
  int kk   = (lane >= 16) ? 8 : 0;
  int k    = (i < 8) ? (32 * t + kk + i) : (32 * t + 16 + kk + (i - 8));
  out[gid] = (_Float16)W[k * Ncols + n];
}

// ---------------- edge attention (WMMA) ----------------
__global__ __launch_bounds__(128)
void attn_kernel(const _Float16* __restrict__ xh, const int* __restrict__ ei,
                 const float* __restrict__ attn_b1,
                 const float* __restrict__ attn_w2,
                 const float* __restrict__ attn_b2,
                 const _Float16* __restrict__ w1p,
                 float* __restrict__ scores, float* __restrict__ sums, int E) {
  __shared__ _Float16 efs[4][16 * 136];  // per-wave 16x128 ef tile, padded
  __shared__ float bsum[HEADS];
  const int lane = threadIdx.x & 31;
  const int wave = threadIdx.x >> 5;
  if (threadIdx.x < HEADS) bsum[threadIdx.x] = 0.f;
  const int ebase = (blockIdx.x * 4 + wave) * 16;

  // gather ef = [x[row] || x[col]] as f16, 16B chunks
  for (int idx = lane; idx < 16 * 16; idx += 32) {
    int e  = idx >> 4;
    int c  = idx & 15;
    int k0 = c * 8;
    int ge = ebase + e;
    int src = (k0 < IN) ? ei[ge] : ei[E + ge];
    int kk  = (k0 < IN) ? k0 : (k0 - IN);
    *(float4*)(&efs[wave][e * 136 + k0]) =
        *(const float4*)(xh + (long)src * IN + kk);
  }
  __syncthreads();

  const int nlo    = lane & 15;
  const int half16 = (lane >= 16) ? 8 : 0;

  for (int h = 0; h < HEADS; ++h) {
    float ps[8];
#pragma unroll
    for (int r = 0; r < 8; ++r) ps[r] = 0.f;
    const _Float16* wp = w1p + h * (4 * 8 * 32 * 16);
#pragma unroll
    for (int j = 0; j < 8; ++j) {  // N-tiles of hidden (HID=128)
      v8f c = {};
#pragma unroll
      for (int t = 0; t < 4; ++t) {  // K=128 in steps of 32
        V16 a;
        const _Float16* ab = &efs[wave][nlo * 136 + 32 * t + half16];
        a.p.lo = *(const v8h*)(ab);
        a.p.hi = *(const v8h*)(ab + 16);
        v16h b = *(const v16h*)(wp + ((t * 8 + j) * 32 + lane) * 16);
        c = __builtin_amdgcn_wmma_f32_16x16x32_f16(false, a.v, false, b,
                                                   (short)0, c, false, false);
      }
      int n = j * 16 + nlo;
      float b1n = attn_b1[h * HID + n];
      float w2n = attn_w2[h * HID + n];
#pragma unroll
      for (int r = 0; r < 8; ++r) ps[r] += fast_tanh(c[r] + b1n) * w2n;
    }
    // reduce over the 16 lanes sharing each M-row set
#pragma unroll
    for (int off = 1; off < 16; off <<= 1) {
#pragma unroll
      for (int r = 0; r < 8; ++r) ps[r] += __shfl_xor(ps[r], off, 32);
    }
    if (nlo == 0) {
      float b2h = attn_b2[h];
      float es = 0.f;
#pragma unroll
      for (int r = 0; r < 8; ++r) {
        float sc = ps[r] + b2h;
        scores[(long)(ebase + half16 + r) * HEADS + h] = sc;
        es += __expf(sc);
      }
      __hip_atomic_fetch_add(&bsum[h], es, __ATOMIC_RELAXED,
                             __HIP_MEMORY_SCOPE_WORKGROUP);
    }
  }
  __syncthreads();
  if (threadIdx.x < HEADS)
    __hip_atomic_fetch_add(&sums[threadIdx.x], bsum[threadIdx.x],
                           __ATOMIC_RELAXED, __HIP_MEMORY_SCOPE_AGENT);
}

// ---------------- softmax normalize (in place) ----------------
__global__ void wnorm_kernel(float* __restrict__ scores,
                             const float* __restrict__ sums, int EH) {
  int gid = blockIdx.x * blockDim.x + threadIdx.x;
  if (gid >= EH) return;
  scores[gid] = __expf(scores[gid]) / sums[gid & (HEADS - 1)];
}

// ---------------- weighted scatter-add (segment_sum) ----------------
__global__ void scatter_kernel(const float* __restrict__ x,
                               const int* __restrict__ ei,
                               const float* __restrict__ w,
                               float* __restrict__ agg, int E) {
  long gid = (long)blockIdx.x * blockDim.x + threadIdx.x;
  int e = (int)(gid >> 6);
  int k = (int)(gid & 63);
  if (e >= E) return;
  int row = ei[e], col = ei[E + e];
  float xv = x[(long)col * IN + k];
  float* dst = agg + (long)row * (HEADS * IN) + k;
#pragma unroll
  for (int h = 0; h < HEADS; ++h)
    __hip_atomic_fetch_add(dst + h * IN, w[(long)e * HEADS + h] * xv,
                           __ATOMIC_RELAXED, __HIP_MEMORY_SCOPE_AGENT);
}

// ---------------- fused node MLP (WMMA) ----------------
__global__ __launch_bounds__(128)
void node_kernel(const float* __restrict__ agg, const _Float16* __restrict__ w0p,
                 const _Float16* __restrict__ w1p, const float* __restrict__ b0,
                 const float* __restrict__ b1, const float* __restrict__ bn_g,
                 const float* __restrict__ bn_b, const float* __restrict__ bn_m,
                 const float* __restrict__ bn_v, const float* __restrict__ cls_w,
                 const float* __restrict__ cls_b, float* __restrict__ out, int N) {
  __shared__ _Float16 ag[4][16 * 264];  // 16x256 f16, padded
  __shared__ _Float16 h1[4][16 * 136];  // 16x128 f16, padded
  const int lane = threadIdx.x & 31;
  const int wave = threadIdx.x >> 5;
  const int nbase = (blockIdx.x * 4 + wave) * 16;
  const int nlo = lane & 15;
  const int half16 = (lane >= 16) ? 8 : 0;

  for (int idx = lane; idx < 16 * 64; idx += 32) {
    int m = idx >> 6, c4 = idx & 63;
    int node = nbase + m;
    if (node >= N) node = N - 1;
    float4 v = *(const float4*)(agg + (long)node * 256 + c4 * 4);
    _Float16* d = &ag[wave][m * 264 + c4 * 4];
    d[0] = (_Float16)v.x; d[1] = (_Float16)v.y;
    d[2] = (_Float16)v.z; d[3] = (_Float16)v.w;
  }
  __syncthreads();

  // layer 0: [16x256] @ [256x128]
  v8f acc[8];
#pragma unroll
  for (int j = 0; j < 8; ++j) acc[j] = (v8f){};
#pragma unroll
  for (int t = 0; t < 8; ++t) {
    V16 a;
    const _Float16* ab = &ag[wave][nlo * 264 + 32 * t + half16];
    a.p.lo = *(const v8h*)(ab);
    a.p.hi = *(const v8h*)(ab + 16);
#pragma unroll
    for (int j = 0; j < 8; ++j) {
      v16h b = *(const v16h*)(w0p + ((t * 8 + j) * 32 + lane) * 16);
      acc[j] = __builtin_amdgcn_wmma_f32_16x16x32_f16(false, a.v, false, b,
                                                      (short)0, acc[j], false, false);
    }
  }
#pragma unroll
  for (int j = 0; j < 8; ++j) {
    int n = j * 16 + nlo;
    float sc = bn_g[n] * rsqrtf(bn_v[n] + EPS);
    float sh = bn_b[n] - bn_m[n] * sc;
    float bias = b0[n];
#pragma unroll
    for (int r = 0; r < 8; ++r) {
      float v = (acc[j][r] + bias) * sc + sh;
      h1[wave][(half16 + r) * 136 + n] = (_Float16)(v > 0.f ? v : 0.f);
    }
  }
  __syncthreads();

  // layer 1: [16x128] @ [128x128]
  v8f acc2[8];
#pragma unroll
  for (int j = 0; j < 8; ++j) acc2[j] = (v8f){};
#pragma unroll
  for (int t = 0; t < 4; ++t) {
    V16 a;
    const _Float16* ab = &h1[wave][nlo * 136 + 32 * t + half16];
    a.p.lo = *(const v8h*)(ab);
    a.p.hi = *(const v8h*)(ab + 16);
#pragma unroll
    for (int j = 0; j < 8; ++j) {
      v16h b = *(const v16h*)(w1p + ((t * 8 + j) * 32 + lane) * 16);
      acc2[j] = __builtin_amdgcn_wmma_f32_16x16x32_f16(false, a.v, false, b,
                                                       (short)0, acc2[j], false, false);
    }
  }
  // BN+relu + classifier [128 x 2] epilogue
  float p0[8], p1[8];
#pragma unroll
  for (int r = 0; r < 8; ++r) { p0[r] = 0.f; p1[r] = 0.f; }
#pragma unroll
  for (int j = 0; j < 8; ++j) {
    int n = j * 16 + nlo;
    float sc = bn_g[HID + n] * rsqrtf(bn_v[HID + n] + EPS);
    float sh = bn_b[HID + n] - bn_m[HID + n] * sc;
    float bias = b1[n];
    float cw0 = cls_w[n * 2], cw1 = cls_w[n * 2 + 1];
#pragma unroll
    for (int r = 0; r < 8; ++r) {
      float v = (acc2[j][r] + bias) * sc + sh;
      v = v > 0.f ? v : 0.f;
      p0[r] += v * cw0;
      p1[r] += v * cw1;
    }
  }
#pragma unroll
  for (int off = 1; off < 16; off <<= 1) {
#pragma unroll
    for (int r = 0; r < 8; ++r) {
      p0[r] += __shfl_xor(p0[r], off, 32);
      p1[r] += __shfl_xor(p1[r], off, 32);
    }
  }
  if (nlo == 0) {
#pragma unroll
    for (int r = 0; r < 8; ++r) {
      int node = nbase + half16 + r;
      if (node < N) {
        out[node * 2 + 0] = p0[r] + cls_b[0];
        out[node * 2 + 1] = p1[r] + cls_b[1];
      }
    }
  }
}

// ---------------- host ----------------
extern "C" void kernel_launch(void* const* d_in, const int* in_sizes, int n_in,
                              void* d_out, int out_size, void* d_ws, size_t ws_size,
                              hipStream_t stream) {
  const float* x        = (const float*)d_in[0];
  const int*   ei       = (const int*)d_in[1];
  const float* attn_w1  = (const float*)d_in[2];
  const float* attn_b1  = (const float*)d_in[3];
  const float* attn_w2  = (const float*)d_in[4];
  const float* attn_b2  = (const float*)d_in[5];
  const float* conv_w0  = (const float*)d_in[6];
  const float* conv_b0  = (const float*)d_in[7];
  const float* conv_w1  = (const float*)d_in[8];
  const float* conv_b1  = (const float*)d_in[9];
  const float* bn_gamma = (const float*)d_in[10];
  const float* bn_beta  = (const float*)d_in[11];
  const float* bn_mean  = (const float*)d_in[12];
  const float* bn_var   = (const float*)d_in[13];
  const float* cls_w    = (const float*)d_in[14];
  const float* cls_b    = (const float*)d_in[15];
  float* out = (float*)d_out;

  const int N = in_sizes[0] / IN;
  const int E = in_sizes[1] / 2;

  // workspace layout (bytes, 256-aligned): ~70.6 MB total
  char* ws = (char*)d_ws;
  size_t off = 0;
  _Float16* xh    = (_Float16*)(ws + off); off += (size_t)N * IN * 2;          // 6.4 MB
  off = (off + 255) & ~(size_t)255;
  _Float16* w1ap  = (_Float16*)(ws + off); off += (size_t)HEADS * 16384 * 2;   // 128 KB
  off = (off + 255) & ~(size_t)255;
  _Float16* w0p   = (_Float16*)(ws + off); off += (size_t)32768 * 2;           // 64 KB
  off = (off + 255) & ~(size_t)255;
  _Float16* w1p   = (_Float16*)(ws + off); off += (size_t)16384 * 2;           // 32 KB
  off = (off + 255) & ~(size_t)255;
  float* sums     = (float*)(ws + off);    off += 256;
  float* scores   = (float*)(ws + off);    off += (size_t)E * HEADS * 4;       // 12.8 MB
  off = (off + 255) & ~(size_t)255;
  float* agg      = (float*)(ws + off);    off += (size_t)N * HEADS * IN * 4;  // 51.2 MB

  hipMemsetAsync(agg, 0, (size_t)N * HEADS * IN * 4, stream);
  hipMemsetAsync(sums, 0, HEADS * 4, stream);

  cvt_f16_kernel<<<(N * IN + 255) / 256, 256, 0, stream>>>(x, xh, N * IN);
  for (int h = 0; h < HEADS; ++h)
    pack_b_kernel<<<(16384 + 255) / 256, 256, 0, stream>>>(
        attn_w1 + (size_t)h * 128 * 128, w1ap + (size_t)h * 16384, 8, 128, 16384);
  pack_b_kernel<<<(32768 + 255) / 256, 256, 0, stream>>>(conv_w0, w0p, 8, 128, 32768);
  pack_b_kernel<<<(16384 + 255) / 256, 256, 0, stream>>>(conv_w1, w1p, 8, 128, 16384);

  attn_kernel<<<E / 64, 128, 0, stream>>>(xh, ei, attn_b1, attn_w2, attn_b2,
                                          w1ap, scores, sums, E);
  wnorm_kernel<<<(E * HEADS + 255) / 256, 256, 0, stream>>>(scores, sums, E * HEADS);
  scatter_kernel<<<((size_t)E * IN + 255) / 256, 256, 0, stream>>>(x, ei, scores, agg, E);
  node_kernel<<<(N + 63) / 64, 128, 0, stream>>>(agg, w0p, w1p, conv_b0, conv_b1,
                                                 bn_gamma, bn_beta, bn_mean, bn_var,
                                                 cls_w, cls_b, out, N);
}
